// MinamoScoreHead_1915555414123
// MI455X (gfx1250) — matrix operand order repaired and checked by reading, not compiled, not run on hardware
//
#include <hip/hip_runtime.h>

typedef __attribute__((ext_vector_type(16))) _Float16 v16h;
typedef __attribute__((ext_vector_type(8)))  _Float16 v8h;
typedef __attribute__((ext_vector_type(8)))  float    v8f;

#define N_NODES 20000
#define N_EDGES 640000
#define DIMC    128
#define OUTC    256
#define NB      64
#define HWD     64
#define OH      62
#define NPIX    (OH*OH)      /* 3844 */
#define KCONV   (DIMC*9)     /* 1152 */
#define NT64    61           /* ceil(3844/64) */
#define NEG     0.2f

__device__ __forceinline__ float leakyf(float x) { return x >= 0.f ? x : NEG * x; }

// float atomic max via int/uint punning (valid for values initialized to -inf)
__device__ __forceinline__ void atomicMaxF(float* a, float v) {
    if (__float_as_int(v) >= 0) atomicMax((int*)a, __float_as_int(v));
    else                        atomicMin((unsigned int*)a, __float_as_uint(v));
}

// ---------------- spectral norm: one power iteration, writes 1/sigma ----------------
__global__ void k_sigma(const float* __restrict__ W, const float* __restrict__ u,
                        float* __restrict__ rsig, int M, int K) {
    __shared__ float vbuf[1152];
    __shared__ float red[256];
    int tid = threadIdx.x;
    for (int j = tid; j < K; j += 256) {
        float s = 0.f;
        for (int i = 0; i < M; ++i) s += W[i * K + j] * u[i];
        vbuf[j] = s;
    }
    __syncthreads();
    float p = 0.f;
    for (int j = tid; j < K; j += 256) p += vbuf[j] * vbuf[j];
    red[tid] = p; __syncthreads();
    for (int s = 128; s > 0; s >>= 1) { if (tid < s) red[tid] += red[tid + s]; __syncthreads(); }
    float inv = 1.f / (sqrtf(red[0]) + 1e-12f);
    __syncthreads();
    for (int j = tid; j < K; j += 256) vbuf[j] *= inv;
    __syncthreads();
    p = 0.f;
    for (int i = tid; i < M; i += 256) {
        float s = 0.f;
        for (int j = 0; j < K; ++j) s += W[i * K + j] * vbuf[j];
        p += s * s;
    }
    red[tid] = p; __syncthreads();
    for (int s = 128; s > 0; s >>= 1) { if (tid < s) red[tid] += red[tid + s]; __syncthreads(); }
    if (tid == 0) {
        float s2 = red[0];
        float sn = sqrtf(s2);
        rsig[0] = (sn + 1e-12f) / fmaxf(s2, 1e-30f);   // 1/sigma, sigma = s2/(sn+eps)
    }
}

// ------------- pack weights into per-lane WMMA B-fragment order (f16) -------------
// frag[((kt*8+nt)*32+lane)*16 + j]  holds  B[kt*32 + (lane>=16?8:0) + (j&7) + (j>=8?16:0)][nt*16 + lane%16]
__global__ void k_pack_conv(const float* __restrict__ w, _Float16* __restrict__ frag) {
    int idx = blockIdx.x * blockDim.x + threadIdx.x;
    if (idx >= 36 * 8 * 32 * 16) return;
    int j    = idx & 15;
    int lane = (idx >> 4) & 31;
    int nt   = (idx >> 9) & 7;
    int kt   = idx >> 12;
    int k = kt * 32 + ((lane >= 16) ? 8 : 0) + (j & 7) + ((j >= 8) ? 16 : 0);
    int n = nt * 16 + (lane & 15);
    frag[idx] = (_Float16)w[n * KCONV + k];         // conv_w[o][i][kh][kw], k=i*9+kh*3+kw
}
__global__ void k_pack_gcn(const float* __restrict__ w, _Float16* __restrict__ frag) {
    int idx = blockIdx.x * blockDim.x + threadIdx.x;
    if (idx >= 4 * 8 * 32 * 16) return;
    int j    = idx & 15;
    int lane = (idx >> 4) & 31;
    int nt   = (idx >> 9) & 7;
    int kt   = idx >> 12;
    int k = kt * 32 + ((lane >= 16) ? 8 : 0) + (j & 7) + ((j >= 8) ? 16 : 0);
    int n = nt * 16 + (lane & 15);
    frag[idx] = (_Float16)w[k * DIMC + n];          // gcn_w[in][out]
}

// ---------------- small utility kernels ----------------
__global__ void k_fill(float* __restrict__ p, float v, int n) {
    int i = blockIdx.x * blockDim.x + threadIdx.x;
    if (i < n) p[i] = v;
}
__global__ void k_init_h(float* __restrict__ h, const float* __restrict__ b) {
    int i = blockIdx.x * blockDim.x + threadIdx.x;
    if (i < N_NODES * DIMC) h[i] = b[i & (DIMC - 1)];
}

// ---------------- vision: implicit-GEMM conv3x3 + leaky + fused 2x2 max-pool ----------------
// grid: 64 images * 61 pixel tiles (64 pixels each); block: 256 = 8 waves.
// Each wave owns one 16-channel N-tile and 4 M-subtiles (4 accumulators) so the
// B fragment loaded per K-step is reused 4x. LDS A-block is double-buffered, and
// staging is split into load (clause of 8 global_load_b32, issued before the
// WMMAs) and convert+ds_store (after the WMMAs) so load latency hides under the
// matrix ops instead of serializing per element.
__global__ void __launch_bounds__(256) k_conv(const float* __restrict__ vis,
                                              const _Float16* __restrict__ frag,
                                              const float* __restrict__ bias,
                                              const float* __restrict__ rsig,
                                              float* __restrict__ pooled) {
    __shared__ __align__(32) _Float16 As[2][64 * 32];
    int tid = threadIdx.x, lane = tid & 31, wave = tid >> 5;
    int img = blockIdx.x / NT64, tile = blockIdx.x % NT64;
    int pbase = tile * 64;

    // staging geometry: this thread owns pixel `spix` for the whole K loop and
    // 8 K-lanes {a, a+4, ..., a+28}. Validity and base pointer hoisted out.
    int spix = ((wave & 1) << 5) | lane;          // 0..63
    int a    = wave >> 1;                         // 0..3
    int ps   = pbase + spix;
    bool valid = ps < NPIX;
    int pc = valid ? ps : 0;                      // clamp -> always in-bounds loads
    int ys = pc / OH, xs = pc - ys * OH;
    const float* base = vis + ((size_t)img * DIMC * HWD + ys) * HWD + xs;

    auto stage_load = [&](int kt, float (&tmp)[8]) {
        #pragma unroll
        for (int i = 0; i < 8; ++i) {
            int k  = kt * 32 + a + 4 * i;
            int ic = k / 9, rem = k - ic * 9;
            int kh = rem / 3, kw = rem - kh * 3;
            tmp[i] = base[ic * (HWD * HWD) + kh * HWD + kw];   // 8 independent loads
        }
    };
    auto stage_store = [&](int buf, const float (&tmp)[8]) {
        #pragma unroll
        for (int i = 0; i < 8; ++i) {
            int kl = a + 4 * i;
            As[buf][spix * 32 + kl] = (_Float16)(valid ? tmp[i] : 0.f);
        }
    };

    v8f acc[4] = {};
    int r = lane & 15;
    int o = (lane < 16) ? 0 : 8;

    {
        float t0[8];
        stage_load(0, t0);
        stage_store(0, t0);
    }
    __syncthreads();
    for (int kt = 0; kt < 36; ++kt) {
        int buf = kt & 1;
        v16h bm = *(const v16h*)(frag + ((size_t)(kt * 8 + wave) * 32 + lane) * 16);
        float tnext[8];
        bool more = (kt + 1 < 36);
        if (more) {
            __builtin_prefetch(frag + ((size_t)((kt + 1) * 8 + wave) * 32 + lane) * 16, 0, 3);
            stage_load(kt + 1, tnext);            // issue loads, consume after WMMAs
        }
        #pragma unroll
        for (int mi = 0; mi < 4; ++mi) {
            const _Float16* ap = &As[buf][(mi * 16 + r) * 32];
            v8h a0 = *(const v8h*)(ap + o);
            v8h a1 = *(const v8h*)(ap + 16 + o);
            v16h av;
            #pragma unroll
            for (int i = 0; i < 8; ++i) { av[i] = a0[i]; av[i + 8] = a1[i]; }
            acc[mi] = __builtin_amdgcn_wmma_f32_16x16x32_f16(false, av, false, bm, (short)0,
                                                             acc[mi], false, false);
        }
        if (more) stage_store(buf ^ 1, tnext);    // convert + ds_store after matrix ops
        __syncthreads();
    }

    // epilogue: 1/sigma scale, bias, leaky, per-quadrant max, atomic-max into pooled[B][128][4]
    float rs = rsig[0];
    int n = wave * 16 + (lane & 15);
    float bi = bias[n];
    float qm[4] = { -__builtin_inff(), -__builtin_inff(), -__builtin_inff(), -__builtin_inff() };
    #pragma unroll
    for (int mi = 0; mi < 4; ++mi) {
        #pragma unroll
        for (int i = 0; i < 8; ++i) {
            int p = pbase + mi * 16 + i + ((lane < 16) ? 0 : 8);
            if (p < NPIX) {
                float v = leakyf(acc[mi][i] * rs + bi);
                int y = p / OH, x = p - y * OH;
                int q = ((y >= 31) ? 2 : 0) + ((x >= 31) ? 1 : 0);
                qm[q] = fmaxf(qm[q], v);
            }
        }
    }
    #pragma unroll
    for (int q = 0; q < 4; ++q)
        if (qm[q] > -__builtin_inff())
            atomicMaxF(&pooled[(img * DIMC + n) * 4 + q], qm[q]);
}

// ---------------- topo: xw = topo @ gcn_w via WMMA ----------------
__global__ void __launch_bounds__(256) k_xw(const float* __restrict__ topo,
                                            const _Float16* __restrict__ frag,
                                            float* __restrict__ xw) {
    __shared__ __align__(32) _Float16 As[16 * 32];
    int tid = threadIdx.x, lane = tid & 31, wave = tid >> 5;
    int rbase = blockIdx.x * 16;
    v8f acc = {};
    int r = lane & 15;
    int o = (lane < 16) ? 0 : 8;
    // two staged elements per thread, loads batched then stores
    int e0 = tid, e1 = tid + 256;
    int kl0 = e0 & 31, px0 = e0 >> 5, kl1 = e1 & 31, px1 = e1 >> 5;
    int row0 = rbase + px0, row1 = rbase + px1;
    bool v0 = row0 < N_NODES, v1 = row1 < N_NODES;
    int r0 = v0 ? row0 : 0, r1 = v1 ? row1 : 0;
    for (int kt = 0; kt < 4; ++kt) {
        float t0 = topo[(size_t)r0 * DIMC + kt * 32 + kl0];
        float t1 = topo[(size_t)r1 * DIMC + kt * 32 + kl1];
        As[px0 * 32 + kl0] = (_Float16)(v0 ? t0 : 0.f);
        As[px1 * 32 + kl1] = (_Float16)(v1 ? t1 : 0.f);
        __syncthreads();
        v8h a0 = *(const v8h*)(As + r * 32 + o);
        v8h a1 = *(const v8h*)(As + r * 32 + 16 + o);
        v16h av;
        #pragma unroll
        for (int i = 0; i < 8; ++i) { av[i] = a0[i]; av[i + 8] = a1[i]; }
        v16h bm = *(const v16h*)(frag + ((size_t)(kt * 8 + wave) * 32 + lane) * 16);
        acc = __builtin_amdgcn_wmma_f32_16x16x32_f16(false, av, false, bm, (short)0, acc,
                                                     false, false);
        __syncthreads();
    }
    int n = wave * 16 + (lane & 15);
    #pragma unroll
    for (int i = 0; i < 8; ++i) {
        int row = rbase + i + ((lane < 16) ? 0 : 8);
        if (row < N_NODES) xw[(size_t)row * DIMC + n] = acc[i];
    }
}

// ---------------- degree (with self loop), sym norm ----------------
__global__ void k_deg(const long long* __restrict__ ei, float* __restrict__ deg) {
    int i = blockIdx.x * blockDim.x + threadIdx.x;
    if (i < N_EDGES) atomicAdd(&deg[(int)ei[N_EDGES + i]], 1.0f);
}
__global__ void k_dis(const float* __restrict__ deg, float* __restrict__ dis) {
    int i = blockIdx.x * blockDim.x + threadIdx.x;
    if (i < N_NODES) dis[i] = rsqrtf(deg[i]);   // deg >= 1 always
}

// ---------------- scatter-add aggregation: one wave per (edge|self-loop) ----------------
__global__ void k_agg(const long long* __restrict__ ei, const float* __restrict__ xw,
                      const float* __restrict__ dis, float* __restrict__ h) {
    int gid = blockIdx.x * blockDim.x + threadIdx.x;
    int w = gid >> 5, lane = gid & 31;
    if (w >= N_EDGES + N_NODES) return;
    int src, dst; float nrm;
    if (w < N_EDGES) {
        src = (int)ei[w]; dst = (int)ei[N_EDGES + w];
        nrm = dis[src] * dis[dst];
    } else {
        src = dst = w - N_EDGES;
        float d = dis[src]; nrm = d * d;
    }
    float4 v = ((const float4*)(xw + (size_t)src * DIMC))[lane];
    float* hp = h + (size_t)dst * DIMC + lane * 4;
    atomicAdd(hp + 0, nrm * v.x);
    atomicAdd(hp + 1, nrm * v.y);
    atomicAdd(hp + 2, nrm * v.z);
    atomicAdd(hp + 3, nrm * v.w);
}

// ---------------- leaky + global max pool per graph ----------------
__global__ void k_pool_t(const float* __restrict__ h, const long long* __restrict__ batch,
                         float* __restrict__ pooled) {
    int gid = blockIdx.x * blockDim.x + threadIdx.x;
    int node = gid >> 5, lane = gid & 31;
    if (node >= N_NODES) return;
    int b = (int)batch[node];
    float4 v = ((const float4*)(h + (size_t)node * DIMC))[lane];
    float* pp = pooled + b * DIMC + lane * 4;
    atomicMaxF(pp + 0, leakyf(v.x));
    atomicMaxF(pp + 1, leakyf(v.y));
    atomicMaxF(pp + 2, leakyf(v.z));
    atomicMaxF(pp + 3, leakyf(v.w));
}

// ---------------- SN linear: out = (feat @ W^T)/sigma + bias ----------------
__global__ void k_fc(const float* __restrict__ feat, const float* __restrict__ W,
                     const float* __restrict__ bias, const float* __restrict__ rsig,
                     float* __restrict__ out, int M, int K, int O) {
    int idx = blockIdx.x * blockDim.x + threadIdx.x;
    if (idx >= M * O) return;
    int m = idx / O, oo = idx - m * O;
    const float4* f = (const float4*)(feat + (size_t)m * K);
    const float4* w = (const float4*)(W + (size_t)oo * K);
    float s = 0.f;
    for (int k = 0; k < (K >> 2); ++k) {
        float4 a = f[k], b = w[k];
        s += a.x * b.x + a.y * b.y + a.z * b.z + a.w * b.w;
    }
    out[idx] = s * rsig[0] + bias[oo];
}

extern "C" void kernel_launch(void* const* d_in, const int* in_sizes, int n_in,
                              void* d_out, int out_size, void* d_ws, size_t ws_size,
                              hipStream_t stream) {
    const float*     vis    = (const float*)d_in[0];
    const float*     topo   = (const float*)d_in[1];
    const long long* ei     = (const long long*)d_in[2];   // int64 [2, E]
    const long long* batch  = (const long long*)d_in[3];   // int64 [N]
    const float*     conv_w = (const float*)d_in[4];
    const float*     conv_b = (const float*)d_in[5];
    const float*     conv_u = (const float*)d_in[6];
    const float*     fcv_w  = (const float*)d_in[7];
    const float*     fcv_b  = (const float*)d_in[8];
    const float*     fcv_u  = (const float*)d_in[9];
    const float*     gcn_w  = (const float*)d_in[10];
    const float*     gcn_b  = (const float*)d_in[11];
    const float*     fct_w  = (const float*)d_in[12];
    const float*     fct_b  = (const float*)d_in[13];
    const float*     fct_u  = (const float*)d_in[14];
    float* out = (float*)d_out;

    // carve workspace (each region 256B aligned), ~21 MB total
    char* ws = (char*)d_ws;
    size_t off = 0;
    auto carve = [&](size_t bytes) -> char* {
        char* p = ws + off;
        off = (off + bytes + 255) & ~(size_t)255;
        return p;
    };
    float*    rsig  = (float*)carve(4 * sizeof(float));
    float*    deg   = (float*)carve((size_t)N_NODES * 4);
    float*    dis   = (float*)carve((size_t)N_NODES * 4);
    float*    xw    = (float*)carve((size_t)N_NODES * DIMC * 4);
    float*    h     = (float*)carve((size_t)N_NODES * DIMC * 4);
    float*    pv    = (float*)carve((size_t)NB * DIMC * 4 * 4);   // [B][128][2][2]
    float*    pt    = (float*)carve((size_t)NB * DIMC * 4);       // [B][128]
    _Float16* fragC = (_Float16*)carve((size_t)36 * 8 * 32 * 16 * 2);
    _Float16* fragG = (_Float16*)carve((size_t)4 * 8 * 32 * 16 * 2);

    // spectral norms (1/sigma)
    k_sigma<<<1, 256, 0, stream>>>(conv_w, conv_u, rsig + 0, DIMC, KCONV);
    k_sigma<<<1, 256, 0, stream>>>(fcv_w,  fcv_u,  rsig + 1, OUTC, DIMC * 4);
    k_sigma<<<1, 256, 0, stream>>>(fct_w,  fct_u,  rsig + 2, OUTC, DIMC);

    // weight fragment packing
    k_pack_conv<<<(36 * 8 * 32 * 16 + 255) / 256, 256, 0, stream>>>(conv_w, fragC);
    k_pack_gcn<<<(4 * 8 * 32 * 16 + 255) / 256, 256, 0, stream>>>(gcn_w, fragG);

    // init accumulators
    k_fill<<<(NB * DIMC * 4 + 255) / 256, 256, 0, stream>>>(pv, -__builtin_inff(), NB * DIMC * 4);
    k_fill<<<(NB * DIMC + 255) / 256, 256, 0, stream>>>(pt, -__builtin_inff(), NB * DIMC);
    k_fill<<<(N_NODES + 255) / 256, 256, 0, stream>>>(deg, 1.0f, N_NODES);   // self loop
    k_init_h<<<(N_NODES * DIMC + 255) / 256, 256, 0, stream>>>(h, gcn_b);

    // topo head
    k_deg<<<(N_EDGES + 255) / 256, 256, 0, stream>>>(ei, deg);
    k_dis<<<(N_NODES + 255) / 256, 256, 0, stream>>>(deg, dis);
    k_xw<<<N_NODES / 16, 256, 0, stream>>>(topo, fragG, xw);
    {
        long long thr = (long long)(N_EDGES + N_NODES) * 32;
        k_agg<<<(int)((thr + 255) / 256), 256, 0, stream>>>(ei, xw, dis, h);
    }
    k_pool_t<<<(N_NODES * 32 + 255) / 256, 256, 0, stream>>>(h, batch, pt);

    // vision head (dominant compute)
    k_conv<<<NB * NT64, 256, 0, stream>>>(vis, fragC, conv_b, rsig, pv);

    // final SN linears -> d_out (vis_score, topo_score)
    k_fc<<<(NB * OUTC + 255) / 256, 256, 0, stream>>>(pv, fcv_w, fcv_b, rsig + 1, out,
                                                      NB, DIMC * 4, OUTC);
    k_fc<<<(NB * OUTC + 255) / 256, 256, 0, stream>>>(pt, fct_w, fct_b, rsig + 2,
                                                      out + NB * OUTC, NB, DIMC, OUTC);
}
